// EncoderProcesserDecoder_15599321219561
// MI455X (gfx1250) — matrix operand order repaired and checked by reading, not compiled, not run on hardware
//
#include <hip/hip_runtime.h>

typedef __bf16 bf16;
typedef __attribute__((ext_vector_type(16))) __bf16 v16bf;
typedef __attribute__((ext_vector_type(8)))  __bf16 v8bf;
typedef __attribute__((ext_vector_type(8)))  float  v8f;
typedef __attribute__((ext_vector_type(4)))  float  v4f;

#define N_NODES 50000
#define N_EDGES 150000
#define HID     128
#define HALFW   64
#define NLAYERS 15
#define FRAG    512   // bf16 elements per swizzled B fragment (32 lanes x 16)

static __device__ __forceinline__ int imin(int a, int b) { return a < b ? a : b; }

static __device__ __forceinline__ float gelu_erf(float x) {
  return 0.5f * x * (1.0f + erff(x * 0.70710678118654752f));
}

// ---------------- WMMA fragment loads ----------------
// A-matrix 16x32 bf16 layout (ISA 7.12.2): lane<16 holds row M=lane, K {8h..8h+7, 8h+16..}
static __device__ __forceinline__ v16bf load_a_frag(const bf16* slab, int stride, int ktile) {
  const int lane = threadIdx.x & 31;
  const int row  = lane & 15;
  const int h    = lane >> 4;
  const bf16* p  = slab + row * stride + ktile * 32 + h * 8;
  v8bf lo = *reinterpret_cast<const v8bf*>(p);
  v8bf hi = *reinterpret_cast<const v8bf*>(p + 16);
  return __builtin_shufflevector(lo, hi, 0,1,2,3,4,5,6,7,8,9,10,11,12,13,14,15);
}

// B fragments pre-swizzled: fragment f = ktile*8+ntile is 32 lanes x 16 bf16 contiguous
static __device__ __forceinline__ v16bf load_b_frag(const bf16* wsw, int ktile, int ntile) {
  const int lane = threadIdx.x & 31;
  const bf16* p  = wsw + (((size_t)(ktile * 8 + ntile)) << 9) + lane * 16;
  v8bf lo = *reinterpret_cast<const v8bf*>(p);
  v8bf hi = *reinterpret_cast<const v8bf*>(p + 8);
  return __builtin_shufflevector(lo, hi, 0,1,2,3,4,5,6,7,8,9,10,11,12,13,14,15);
}

// One wave computes D[16x128] = act(in[16xK] @ W[Kx128] + b); deposits bf16 and/or f32 to LDS
static __device__ __forceinline__ void wave_gemm(
    const bf16* in_slab, int in_stride, int ktiles,
    const bf16* wsw, const float* bias, bool do_gelu,
    bf16* out_bf, float* out_f32)
{
  const int lane = threadIdx.x & 31;
  v8f acc[8];
#pragma unroll
  for (int t = 0; t < 8; ++t)
#pragma unroll
    for (int r = 0; r < 8; ++r) acc[t][r] = 0.0f;

  for (int k = 0; k < ktiles; ++k) {
    v16bf a = load_a_frag(in_slab, in_stride, k);
#pragma unroll
    for (int t = 0; t < 8; ++t) {
      v16bf b = load_b_frag(wsw, k, t);
      acc[t] = __builtin_amdgcn_wmma_f32_16x16x32_bf16(
          false, a, false, b, (short)0, acc[t], false, false);
    }
  }
  // C/D layout: lane covers column n = t*16 + (lane&15); VGPR r -> row r + 8*(lane>>4)
  const int col0 = lane & 15;
  const int rh   = (lane >> 4) * 8;
#pragma unroll
  for (int t = 0; t < 8; ++t) {
    const int n   = t * 16 + col0;
    const float bv = bias[n];
#pragma unroll
    for (int r = 0; r < 8; ++r) {
      float v = acc[t][r] + bv;
      if (do_gelu) v = gelu_erf(v);
      const int row = rh + r;
      if (out_bf)  out_bf[row * HID + n]  = (bf16)v;
      if (out_f32) out_f32[row * HID + n] = v;
    }
  }
}

// LayerNorm over 128 features for a wave's 16x128 f32 slab.
// lane handles row (lane&15), cols [64*(lane>>4), +64); pair-sum via shfl_xor 16.
static __device__ __forceinline__ void wave_layernorm(
    const float* f32slab, const float* g, const float* beta, float* out)
{
  const int lane = threadIdx.x & 31;
  const int row = lane & 15, half = lane >> 4;
  const float* rp = f32slab + row * HID + half * 64;
  float s0 = 0.0f, s1 = 0.0f;
#pragma unroll
  for (int j = 0; j < 64; ++j) { float v = rp[j]; out[j] = v; s0 += v; s1 += v * v; }
  s0 += __shfl_xor(s0, 16);
  s1 += __shfl_xor(s1, 16);
  const float mean = s0 * (1.0f / 128.0f);
  const float var  = s1 * (1.0f / 128.0f) - mean * mean;
  const float rstd = rsqrtf(var + 1e-5f);
#pragma unroll
  for (int j = 0; j < 64; ++j) {
    const int col = half * 64 + j;
    out[j] = (out[j] - mean) * rstd * g[col] + beta[col];
  }
}

// ---------------- weight swizzle: fp32 [Krows x 128] -> bf16 fragment-major ----------------
__global__ void k_swizzle(const float* __restrict__ W, bf16* __restrict__ dst, int Krows) {
  const int f = blockIdx.x;             // ktile*8 + ntile
  const int ktile = f >> 3, ntile = f & 7;
  const int lane = threadIdx.x;         // 32 threads
  const int n = ntile * 16 + (lane & 15);
  const int kbase = ktile * 32 + (lane >> 4) * 8;
  bf16* d = dst + (((size_t)f) << 9) + lane * 16;
#pragma unroll
  for (int e = 0; e < 16; ++e) {
    const int k = kbase + (e & 7) + ((e >> 3) << 4);
    const float v = (k < Krows) ? W[(size_t)k * HID + n] : 0.0f;
    d[e] = (bf16)v;
  }
}

__global__ void k_zero(float* __restrict__ p, int n) {
  const int i = blockIdx.x * 256 + threadIdx.x;
  if (i < n) p[i] = 0.0f;
}

// ---------------- encoder: MLP(16->128->128->128) + LN ----------------
__global__ void __launch_bounds__(64)
k_encode(const float* __restrict__ X, int rows,
         const bf16* W1s, const float* b1, const bf16* W2s, const float* b2,
         const bf16* W3s, const float* b3, const float* g, const float* beta,
         float* __restrict__ out_f32, bf16* __restrict__ out_bf)
{
  __shared__ __align__(16) bf16  in_slab[32 * 32];
  __shared__ __align__(16) bf16  bufA[2 * 16 * HID];
  __shared__ __align__(16) bf16  bufB[2 * 16 * HID];
  __shared__ __align__(16) float f32s[2 * 16 * HID];
  const int tid = threadIdx.x;
  const int base = blockIdx.x * 32;

  for (int it = tid; it < 32 * 16; it += 64) {
    const int rl = it >> 4, c = it & 15;
    const int gr = imin(base + rl, rows - 1);
    in_slab[rl * 32 + c]       = (bf16)X[(size_t)gr * 16 + c];
    in_slab[rl * 32 + 16 + c]  = (bf16)0.0f;   // K pad 16..31
  }
  __syncthreads();

  const int w = tid >> 5, lane = tid & 31;
  bf16*  wbA = bufA + w * 16 * HID;
  bf16*  wbB = bufB + w * 16 * HID;
  float* wf  = f32s + w * 16 * HID;

  wave_gemm(in_slab + w * 16 * 32, 32, 1, W1s, b1, true, wbA, nullptr);
  __syncthreads();
  wave_gemm(wbA, HID, 4, W2s, b2, true, wbB, nullptr);
  __syncthreads();
  wave_gemm(wbB, HID, 4, W3s, b3, false, nullptr, wf);
  __syncthreads();

  float vals[64];
  wave_layernorm(wf, g, beta, vals);
  const int row = lane & 15, half = lane >> 4;
  const int gr = base + w * 16 + row;
  if (gr < rows) {
    float* op = out_f32 + (size_t)gr * HID + half * 64;
#pragma unroll
    for (int j = 0; j < 64; j += 4) {
      v4f t; t[0] = vals[j]; t[1] = vals[j + 1]; t[2] = vals[j + 2]; t[3] = vals[j + 3];
      *reinterpret_cast<v4f*>(op + j) = t;
    }
    if (out_bf) {
      bf16* ob = out_bf + (size_t)gr * HID + half * 64;
#pragma unroll
      for (int j = 0; j < 64; ++j) ob[j] = (bf16)vals[j];
    }
  }
}

// ---------------- edge block: concat gather -> MLP(384..) + LN -> residual + scatter ----------------
__global__ void __launch_bounds__(64)
k_edge_block(const int* __restrict__ senders, const int* __restrict__ receivers,
             const bf16* __restrict__ n_bf, float* __restrict__ e_f32, float* __restrict__ agg,
             const bf16* W1s, const float* b1, const bf16* W2s, const float* b2,
             const bf16* W3s, const float* b3, const float* g, const float* beta)
{
  __shared__ int sidx[32], ridx[32];
  __shared__ __align__(16) bf16  in_slab[32 * 384];
  __shared__ __align__(16) bf16  bufA[2 * 16 * HID];
  __shared__ __align__(16) bf16  bufB[2 * 16 * HID];
  __shared__ __align__(16) float f32s[2 * 16 * HID];
  const int tid = threadIdx.x;
  const int base = blockIdx.x * 32;

  if (tid < 32) {
    const int e = imin(base + tid, N_EDGES - 1);
    sidx[tid] = senders[e];
    ridx[tid] = receivers[e];
  }
  __syncthreads();

  for (int it = tid; it < 32 * 24; it += 64) {
    const int el = it / 24, c = it % 24;   // 24 chunks of 16 cols
    bf16* dst = in_slab + el * 384 + c * 16;
    if (c < 8) {
      const bf16* src = n_bf + (size_t)sidx[el] * HID + c * 16;
      *reinterpret_cast<v8bf*>(dst)     = *reinterpret_cast<const v8bf*>(src);
      *reinterpret_cast<v8bf*>(dst + 8) = *reinterpret_cast<const v8bf*>(src + 8);
    } else if (c < 16) {
      const bf16* src = n_bf + (size_t)ridx[el] * HID + (c - 8) * 16;
      *reinterpret_cast<v8bf*>(dst)     = *reinterpret_cast<const v8bf*>(src);
      *reinterpret_cast<v8bf*>(dst + 8) = *reinterpret_cast<const v8bf*>(src + 8);
    } else {
      const int ge = imin(base + el, N_EDGES - 1);
      const float* src = e_f32 + (size_t)ge * HID + (c - 16) * 16;
#pragma unroll
      for (int q = 0; q < 4; ++q) {
        v4f v = *reinterpret_cast<const v4f*>(src + q * 4);
        dst[q * 4 + 0] = (bf16)v[0]; dst[q * 4 + 1] = (bf16)v[1];
        dst[q * 4 + 2] = (bf16)v[2]; dst[q * 4 + 3] = (bf16)v[3];
      }
    }
  }
  __syncthreads();

  const int w = tid >> 5, lane = tid & 31;
  bf16*  wbA = bufA + w * 16 * HID;
  bf16*  wbB = bufB + w * 16 * HID;
  float* wf  = f32s + w * 16 * HID;

  wave_gemm(in_slab + w * 16 * 384, 384, 12, W1s, b1, true, wbA, nullptr);
  __syncthreads();
  wave_gemm(wbA, HID, 4, W2s, b2, true, wbB, nullptr);
  __syncthreads();
  wave_gemm(wbB, HID, 4, W3s, b3, false, nullptr, wf);
  __syncthreads();

  float vals[64];
  wave_layernorm(wf, g, beta, vals);
  const int row = lane & 15, half = lane >> 4;
  const int el = w * 16 + row;
  const int ge = base + el;
  if (ge < N_EDGES) {
    float* erow = e_f32 + (size_t)ge * HID + half * 64;
    const int tgt = (half == 0) ? ridx[el] : sidx[el];   // cols 0..63 -> receiver, 64..127 -> sender
    float* arow = agg + (size_t)tgt * HALFW;
#pragma unroll
    for (int j = 0; j < 64; j += 4) {
      v4f old = *reinterpret_cast<const v4f*>(erow + j);
      v4f nv;
#pragma unroll
      for (int q = 0; q < 4; ++q) {
        const float en = vals[j + q];
        __hip_atomic_fetch_add(arow + j + q, en, __ATOMIC_RELAXED, __HIP_MEMORY_SCOPE_AGENT);
        nv[q] = old[q] + en;                              // residual: e += e_new
      }
      *reinterpret_cast<v4f*>(erow + j) = nv;
    }
  }
}

// ---------------- node block: concat [x, agg] -> MLP(192..) + LN -> residual ----------------
__global__ void __launch_bounds__(64)
k_node_block(float* __restrict__ n_f32, bf16* __restrict__ n_bf, const float* __restrict__ agg,
             const bf16* W1s, const float* b1, const bf16* W2s, const float* b2,
             const bf16* W3s, const float* b3, const float* g, const float* beta)
{
  __shared__ __align__(16) bf16  in_slab[32 * 192];
  __shared__ __align__(16) bf16  bufA[2 * 16 * HID];
  __shared__ __align__(16) bf16  bufB[2 * 16 * HID];
  __shared__ __align__(16) float f32s[2 * 16 * HID];
  const int tid = threadIdx.x;
  const int base = blockIdx.x * 32;

  for (int it = tid; it < 32 * 12; it += 64) {
    const int rl = it / 12, c = it % 12;
    const int gr = imin(base + rl, N_NODES - 1);
    bf16* dst = in_slab + rl * 192 + c * 16;
    if (c < 8) {
      const bf16* src = n_bf + (size_t)gr * HID + c * 16;
      *reinterpret_cast<v8bf*>(dst)     = *reinterpret_cast<const v8bf*>(src);
      *reinterpret_cast<v8bf*>(dst + 8) = *reinterpret_cast<const v8bf*>(src + 8);
    } else {
      const float* src = agg + (size_t)gr * HALFW + (c - 8) * 16;
#pragma unroll
      for (int q = 0; q < 4; ++q) {
        v4f v = *reinterpret_cast<const v4f*>(src + q * 4);
        dst[q * 4 + 0] = (bf16)v[0]; dst[q * 4 + 1] = (bf16)v[1];
        dst[q * 4 + 2] = (bf16)v[2]; dst[q * 4 + 3] = (bf16)v[3];
      }
    }
  }
  __syncthreads();

  const int w = tid >> 5, lane = tid & 31;
  bf16*  wbA = bufA + w * 16 * HID;
  bf16*  wbB = bufB + w * 16 * HID;
  float* wf  = f32s + w * 16 * HID;

  wave_gemm(in_slab + w * 16 * 192, 192, 6, W1s, b1, true, wbA, nullptr);
  __syncthreads();
  wave_gemm(wbA, HID, 4, W2s, b2, true, wbB, nullptr);
  __syncthreads();
  wave_gemm(wbB, HID, 4, W3s, b3, false, nullptr, wf);
  __syncthreads();

  float vals[64];
  wave_layernorm(wf, g, beta, vals);
  const int row = lane & 15, half = lane >> 4;
  const int gr = base + w * 16 + row;
  if (gr < N_NODES) {
    float* xr = n_f32 + (size_t)gr * HID + half * 64;
    bf16*  xb = n_bf  + (size_t)gr * HID + half * 64;
#pragma unroll
    for (int j = 0; j < 64; j += 4) {
      v4f old = *reinterpret_cast<const v4f*>(xr + j);
      v4f nv;
#pragma unroll
      for (int q = 0; q < 4; ++q) nv[q] = old[q] + vals[j + q];   // residual x += x_new
      *reinterpret_cast<v4f*>(xr + j) = nv;
#pragma unroll
      for (int q = 0; q < 4; ++q) xb[j + q] = (bf16)nv[q];
    }
  }
}

// ---------------- decoder: MLP(128->128->128->3), no LN ----------------
__global__ void __launch_bounds__(64)
k_decode(const bf16* __restrict__ n_bf,
         const bf16* W1s, const float* b1, const bf16* W2s, const float* b2,
         const float* __restrict__ W3, const float* __restrict__ b3,
         float* __restrict__ out)
{
  __shared__ __align__(16) bf16  in_slab[32 * HID];
  __shared__ __align__(16) bf16  bufA[2 * 16 * HID];
  __shared__ __align__(16) float f32s[2 * 16 * HID];
  const int tid = threadIdx.x;
  const int base = blockIdx.x * 32;

  for (int it = tid; it < 32 * 8; it += 64) {
    const int rl = it >> 3, c = it & 7;
    const int gr = imin(base + rl, N_NODES - 1);
    bf16* dst = in_slab + rl * HID + c * 16;
    const bf16* src = n_bf + (size_t)gr * HID + c * 16;
    *reinterpret_cast<v8bf*>(dst)     = *reinterpret_cast<const v8bf*>(src);
    *reinterpret_cast<v8bf*>(dst + 8) = *reinterpret_cast<const v8bf*>(src + 8);
  }
  __syncthreads();

  const int w = tid >> 5, lane = tid & 31;
  wave_gemm(in_slab + w * 16 * HID, HID, 4, W1s, b1, true, bufA + w * 16 * HID, nullptr);
  __syncthreads();
  wave_gemm(bufA + w * 16 * HID, HID, 4, W2s, b2, true, nullptr, f32s + w * 16 * HID);
  __syncthreads();

  // last layer 128 -> 3 as VALU dot products
  const int row = lane & 15, half = lane >> 4;
  const float* hrow = f32s + w * 16 * HID + row * HID + half * 64;
  float p0 = 0.f, p1 = 0.f, p2 = 0.f;
#pragma unroll
  for (int j = 0; j < 64; ++j) {
    const float hv = hrow[j];
    const float* wr = W3 + (size_t)(half * 64 + j) * 3;
    p0 += hv * wr[0]; p1 += hv * wr[1]; p2 += hv * wr[2];
  }
  p0 += __shfl_xor(p0, 16); p1 += __shfl_xor(p1, 16); p2 += __shfl_xor(p2, 16);
  const int gr = base + w * 16 + row;
  if (half == 0 && gr < N_NODES) {
    out[(size_t)gr * 3 + 0] = p0 + b3[0];
    out[(size_t)gr * 3 + 1] = p1 + b3[1];
    out[(size_t)gr * 3 + 2] = p2 + b3[2];
  }
}

// ---------------- host ----------------
extern "C" void kernel_launch(void* const* d_in, const int* in_sizes, int n_in,
                              void* d_out, int out_size, void* d_ws, size_t ws_size,
                              hipStream_t stream)
{
  (void)in_sizes; (void)n_in; (void)out_size; (void)ws_size;
  int i = 0;
  const float* x          = (const float*)d_in[i++];
  const float* edge_attr  = (const float*)d_in[i++];
  const float* enc_nW1 = (const float*)d_in[i++]; const float* enc_nb1 = (const float*)d_in[i++];
  const float* enc_nW2 = (const float*)d_in[i++]; const float* enc_nb2 = (const float*)d_in[i++];
  const float* enc_nW3 = (const float*)d_in[i++]; const float* enc_nb3 = (const float*)d_in[i++];
  const float* enc_ng  = (const float*)d_in[i++]; const float* enc_nbt = (const float*)d_in[i++];
  const float* enc_eW1 = (const float*)d_in[i++]; const float* enc_eb1 = (const float*)d_in[i++];
  const float* enc_eW2 = (const float*)d_in[i++]; const float* enc_eb2 = (const float*)d_in[i++];
  const float* enc_eW3 = (const float*)d_in[i++]; const float* enc_eb3 = (const float*)d_in[i++];
  const float* enc_eg  = (const float*)d_in[i++]; const float* enc_ebt = (const float*)d_in[i++];
  const float* blk_eW1 = (const float*)d_in[i++]; const float* blk_eb1 = (const float*)d_in[i++];
  const float* blk_eW2 = (const float*)d_in[i++]; const float* blk_eb2 = (const float*)d_in[i++];
  const float* blk_eW3 = (const float*)d_in[i++]; const float* blk_eb3 = (const float*)d_in[i++];
  const float* blk_eg  = (const float*)d_in[i++]; const float* blk_ebt = (const float*)d_in[i++];
  const float* blk_nW1 = (const float*)d_in[i++]; const float* blk_nb1 = (const float*)d_in[i++];
  const float* blk_nW2 = (const float*)d_in[i++]; const float* blk_nb2 = (const float*)d_in[i++];
  const float* blk_nW3 = (const float*)d_in[i++]; const float* blk_nb3 = (const float*)d_in[i++];
  const float* blk_ng  = (const float*)d_in[i++]; const float* blk_nbt = (const float*)d_in[i++];
  const float* dec_W1  = (const float*)d_in[i++]; const float* dec_b1  = (const float*)d_in[i++];
  const float* dec_W2  = (const float*)d_in[i++]; const float* dec_b2  = (const float*)d_in[i++];
  const float* dec_W3  = (const float*)d_in[i++]; const float* dec_b3  = (const float*)d_in[i++];
  const int*   eidx    = (const int*)d_in[i++];

  char* wp = (char*)d_ws;
  auto alloc = [&](size_t bytes) -> void* {
    void* r = (void*)wp; wp += (bytes + 255) & ~(size_t)255; return r;
  };
  float* n_f32 = (float*)alloc((size_t)N_NODES * HID * 4);
  float* e_f32 = (float*)alloc((size_t)N_EDGES * HID * 4);
  bf16*  n_bf  = (bf16*) alloc((size_t)N_NODES * HID * 2);
  float* agg   = (float*)alloc((size_t)N_NODES * HALFW * 4);

  auto allocw = [&](int ktiles) -> bf16* { return (bf16*)alloc((size_t)ktiles * 8 * FRAG * 2); };
  bf16* enc_nW1s = allocw(1); bf16* enc_nW2s = allocw(4); bf16* enc_nW3s = allocw(4);
  bf16* enc_eW1s = allocw(1); bf16* enc_eW2s = allocw(4); bf16* enc_eW3s = allocw(4);
  bf16* blk_eW1s = allocw(12 * NLAYERS);
  bf16* blk_eW2s = allocw(4  * NLAYERS);
  bf16* blk_eW3s = allocw(4  * NLAYERS);
  bf16* blk_nW1s = allocw(6  * NLAYERS);
  bf16* blk_nW2s = allocw(4  * NLAYERS);
  bf16* blk_nW3s = allocw(4  * NLAYERS);
  bf16* dec_W1s  = allocw(4); bf16* dec_W2s = allocw(4);

  // weight swizzle (cheap; ~2M elements total)
  k_swizzle<<<1 * 8, 32, 0, stream>>>(enc_nW1, enc_nW1s, 16);
  k_swizzle<<<4 * 8, 32, 0, stream>>>(enc_nW2, enc_nW2s, 128);
  k_swizzle<<<4 * 8, 32, 0, stream>>>(enc_nW3, enc_nW3s, 128);
  k_swizzle<<<1 * 8, 32, 0, stream>>>(enc_eW1, enc_eW1s, 16);
  k_swizzle<<<4 * 8, 32, 0, stream>>>(enc_eW2, enc_eW2s, 128);
  k_swizzle<<<4 * 8, 32, 0, stream>>>(enc_eW3, enc_eW3s, 128);
  for (int l = 0; l < NLAYERS; ++l) {
    k_swizzle<<<12 * 8, 32, 0, stream>>>(blk_eW1 + (size_t)l * 384 * HID, blk_eW1s + (size_t)l * 12 * 8 * FRAG, 384);
    k_swizzle<<<4 * 8, 32, 0, stream>>>(blk_eW2 + (size_t)l * HID * HID, blk_eW2s + (size_t)l * 4 * 8 * FRAG, 128);
    k_swizzle<<<4 * 8, 32, 0, stream>>>(blk_eW3 + (size_t)l * HID * HID, blk_eW3s + (size_t)l * 4 * 8 * FRAG, 128);
    k_swizzle<<<6 * 8, 32, 0, stream>>>(blk_nW1 + (size_t)l * 192 * HID, blk_nW1s + (size_t)l * 6 * 8 * FRAG, 192);
    k_swizzle<<<4 * 8, 32, 0, stream>>>(blk_nW2 + (size_t)l * HID * HID, blk_nW2s + (size_t)l * 4 * 8 * FRAG, 128);
    k_swizzle<<<4 * 8, 32, 0, stream>>>(blk_nW3 + (size_t)l * HID * HID, blk_nW3s + (size_t)l * 4 * 8 * FRAG, 128);
  }
  k_swizzle<<<4 * 8, 32, 0, stream>>>(dec_W1, dec_W1s, 128);
  k_swizzle<<<4 * 8, 32, 0, stream>>>(dec_W2, dec_W2s, 128);

  const int ngrid = (N_NODES + 31) / 32;
  const int egrid = (N_EDGES + 31) / 32;

  k_encode<<<ngrid, 64, 0, stream>>>(x, N_NODES,
      enc_nW1s, enc_nb1, enc_nW2s, enc_nb2, enc_nW3s, enc_nb3, enc_ng, enc_nbt,
      n_f32, n_bf);
  k_encode<<<egrid, 64, 0, stream>>>(edge_attr, N_EDGES,
      enc_eW1s, enc_eb1, enc_eW2s, enc_eb2, enc_eW3s, enc_eb3, enc_eg, enc_ebt,
      e_f32, nullptr);

  for (int l = 0; l < NLAYERS; ++l) {
    k_zero<<<(N_NODES * HALFW + 255) / 256, 256, 0, stream>>>(agg, N_NODES * HALFW);
    k_edge_block<<<egrid, 64, 0, stream>>>(eidx, eidx + N_EDGES, n_bf, e_f32, agg,
        blk_eW1s + (size_t)l * 12 * 8 * FRAG, blk_eb1 + (size_t)l * HID,
        blk_eW2s + (size_t)l * 4  * 8 * FRAG, blk_eb2 + (size_t)l * HID,
        blk_eW3s + (size_t)l * 4  * 8 * FRAG, blk_eb3 + (size_t)l * HID,
        blk_eg + (size_t)l * HID, blk_ebt + (size_t)l * HID);
    k_node_block<<<ngrid, 64, 0, stream>>>(n_f32, n_bf, agg,
        blk_nW1s + (size_t)l * 6 * 8 * FRAG, blk_nb1 + (size_t)l * HID,
        blk_nW2s + (size_t)l * 4 * 8 * FRAG, blk_nb2 + (size_t)l * HID,
        blk_nW3s + (size_t)l * 4 * 8 * FRAG, blk_nb3 + (size_t)l * HID,
        blk_ng + (size_t)l * HID, blk_nbt + (size_t)l * HID);
  }

  k_decode<<<ngrid, 64, 0, stream>>>(n_bf, dec_W1s, dec_b1, dec_W2s, dec_b2,
                                     dec_W3, dec_b3, (float*)d_out);
}
